// SSIMLoss_23278722744856
// MI455X (gfx1250) — matrix operand order repaired and checked by reading, not compiled, not run on hardware
//
#include <hip/hip_runtime.h>
#include <math.h>

typedef __attribute__((ext_vector_type(16))) _Float16 v16h;
typedef __attribute__((ext_vector_type(8)))  float    v8f;

#define IMG_H 512
#define IMG_W 512
#define IMG_PIX (IMG_H * IMG_W)
#define N_IMG 48                          /* 16 batch * 3 channels */
#define TILES_X 32
#define TILES_PER_IMG (TILES_X * TILES_X)
#define N_TILES (N_IMG * TILES_PER_IMG)   /* 49152 */
#define WAVES_PER_BLOCK 8
#define N_BLOCKS (N_TILES / WAVES_PER_BLOCK) /* 6144 */

// ---------------------------------------------------------------------------
// Setup: bake the banded Gaussian weight matrix into a per-lane f16 WMMA
// B-operand table (32x16, B[k][j] = g[k-j] for 0<=k-j<=10 else 0).
// The same table serves the horizontal pass (k = column tap) and the vertical
// pass (k = row tap), because both are the identical band matrix.
// Lane L -> column j = L&15; element e -> k = 8*(L>>4) + e (e<8), 16 + 8*(L>>4) + (e-8) (e>=8).
// g is recovered from the reference's own window: w2d[5][t] = g[5]*g[t].
// ---------------------------------------------------------------------------
__global__ void ssim_setup(const float* __restrict__ window,
                           v16h* __restrict__ btab)
{
    const int lane = threadIdx.x;      // 32 threads = 1 wave
    const int m    = lane & 15;
    const int half = lane >> 4;
    const float g5 = sqrtf(window[5 * 11 + 5]);
    v16h b;
    #pragma unroll
    for (int e = 0; e < 16; ++e) {
        int k = (e < 8) ? (8 * half + e) : (16 + 8 * half + (e - 8));
        int d = k - m;
        float w = (d >= 0 && d <= 10) ? (window[5 * 11 + d] / g5) : 0.0f;
        b[e] = (_Float16)w;
    }
    btab[lane] = b;
}

// ---------------------------------------------------------------------------
// One wave per 16x16 SSIM-map tile. Per quantity q in {p, t, p^2, t^2, p*t}:
//   h1 = A1 x Bh   (rows s=0..15 of the 26-row horizontal conv strip)
//   h2 = A2 x Bh   (rows s=16..25; s=26..31 dead, killed by zero weights)
//   acc = [h1|h2] x Bh   -- the D-layout of h1/h2 IS the A-layout of H^T,
//                           so the vertical pass chains with no shuffles and
//                           produces out^T (layout-consistent across all 5).
// 15 v_wmma_f32_16x16x32_f16 per tile.
// ---------------------------------------------------------------------------
__global__ __launch_bounds__(256) void ssim_tile_kernel(
    const float* __restrict__ pred,
    const float* __restrict__ target,
    const v16h* __restrict__ btab,
    float* __restrict__ partials)
{
    const int lane = threadIdx.x & 31;
    const int wave = threadIdx.x >> 5;
    const int tile = blockIdx.x * WAVES_PER_BLOCK + wave;
    const int img  = tile / TILES_PER_IMG;
    const int trem = tile - img * TILES_PER_IMG;
    const int ty   = trem >> 5;
    const int tx   = trem & 31;
    const int row0 = ty * 16;
    const int col0 = tx * 16;

    const int m    = lane & 15;
    const int half = lane >> 4;

    const v16h bh = btab[lane];

    const float* P = pred   + (size_t)img * IMG_PIX;
    const float* T = target + (size_t)img * IMG_PIX;
    const int r1 = row0 - 5 + m;        // strip rows s = 0..15
    const int r2 = row0 + 11 + m;       // strip rows s = 16..31

    float p1[16], t1[16], p2[16], t2[16];

    // Interior test is wave-uniform (row0/col0 uniform per wave): scalar
    // branch, EXEC stays all-ones on both sides (WMMA requirement).
    const bool interior = (row0 >= 16) && (row0 <= 480) &&
                          (col0 >= 16) && (col0 <= 480);
    if (interior) {
        const float* Pr1 = P + r1 * IMG_W;
        const float* Tr1 = T + r1 * IMG_W;
        const float* Pr2 = P + r2 * IMG_W;   // r2 <= 506: in bounds; rows
        const float* Tr2 = T + r2 * IMG_W;   // s>=26 are zero-weighted.
        #pragma unroll
        for (int e = 0; e < 16; ++e) {
            int c = col0 - 5 + ((e < 8) ? (8 * half + e)
                                        : (16 + 8 * half + (e - 8)));
            p1[e] = Pr1[c];
            t1[e] = Tr1[c];
            p2[e] = Pr2[c];
            t2[e] = Tr2[c];
        }
    } else {
        const bool row1ok = (r1 >= 0) && (r1 < IMG_H);
        const bool row2ok = (m < 10) && (r2 < IMG_H);
        #pragma unroll
        for (int e = 0; e < 16; ++e) {
            int c = col0 - 5 + ((e < 8) ? (8 * half + e)
                                        : (16 + 8 * half + (e - 8)));
            bool cok = (c >= 0) && (c < IMG_W);
            int i1 = r1 * IMG_W + c;
            int i2 = r2 * IMG_W + c;
            bool ok1 = row1ok && cok;
            bool ok2 = row2ok && cok;
            p1[e] = ok1 ? P[i1] : 0.0f;
            t1[e] = ok1 ? T[i1] : 0.0f;
            p2[e] = ok2 ? P[i2] : 0.0f;
            t2[e] = ok2 ? T[i2] : 0.0f;
        }
    }

    const v8f zc = {0.f, 0.f, 0.f, 0.f, 0.f, 0.f, 0.f, 0.f};
    v8f acc[5];

    #pragma unroll
    for (int q = 0; q < 5; ++q) {
        v16h a1, a2;
        #pragma unroll
        for (int e = 0; e < 16; ++e) {
            float x1, x2;
            switch (q) {
                case 0:  x1 = p1[e];          x2 = p2[e];          break;
                case 1:  x1 = t1[e];          x2 = t2[e];          break;
                case 2:  x1 = p1[e] * p1[e];  x2 = p2[e] * p2[e];  break;
                case 3:  x1 = t1[e] * t1[e];  x2 = t2[e] * t2[e];  break;
                default: x1 = p1[e] * t1[e];  x2 = p2[e] * t2[e];  break;
            }
            a1[e] = (_Float16)x1;
            a2[e] = (_Float16)x2;
        }

        v8f h1 = __builtin_amdgcn_wmma_f32_16x16x32_f16(
            false, a1, false, bh, (short)0, zc, false, false);
        v8f h2 = __builtin_amdgcn_wmma_f32_16x16x32_f16(
            false, a2, false, bh, (short)0, zc, false, false);

        // D-layout -> A-layout chaining: lane L's h1[v] is H[v+8*half][L&15],
        // which is exactly A-element K = 8*half+v of row (L&15) of H^T;
        // h2[v] supplies K = 16+8*half+v. No cross-lane traffic.
        v16h a3;
        #pragma unroll
        for (int e = 0; e < 8; ++e) {
            a3[e]     = (_Float16)h1[e];
            a3[e + 8] = (_Float16)h2[e];
        }

        acc[q] = __builtin_amdgcn_wmma_f32_16x16x32_f16(
            false, a3, false, bh, (short)0, zc, false, false);
    }

    // ---- Elementwise SSIM (all 5 accs share the same out^T layout) ----
    const float C1 = 1e-4f;   // 0.01^2
    const float C2 = 9e-4f;   // 0.03^2
    float lsum = 0.0f;
    #pragma unroll
    for (int v = 0; v < 8; ++v) {
        float mu1 = acc[0][v], mu2 = acc[1][v];
        float e11 = acc[2][v], e22 = acc[3][v], e12 = acc[4][v];
        float mu1sq = mu1 * mu1, mu2sq = mu2 * mu2, mu12 = mu1 * mu2;
        float s1  = e11 - mu1sq;
        float s2  = e22 - mu2sq;
        float s12 = e12 - mu12;
        float num = (2.0f * mu12 + C1) * (2.0f * s12 + C2);
        float den = (mu1sq + mu2sq + C1) * (s1 + s2 + C2);
        lsum += num / den;
    }

    // Wave32 reduction, then per-block partial (deterministic order).
    #pragma unroll
    for (int off = 16; off >= 1; off >>= 1) lsum += __shfl_xor(lsum, off);

    __shared__ float wsum[WAVES_PER_BLOCK];
    if (lane == 0) wsum[wave] = lsum;
    __syncthreads();
    if (threadIdx.x == 0) {
        float b = 0.0f;
        #pragma unroll
        for (int i = 0; i < WAVES_PER_BLOCK; ++i) b += wsum[i];
        partials[blockIdx.x] = b;
    }
}

__global__ __launch_bounds__(256) void ssim_finalize(
    const float* __restrict__ partials, float* __restrict__ out)
{
    __shared__ float s[256];
    float a = 0.0f;
    for (int i = threadIdx.x; i < N_BLOCKS; i += 256) a += partials[i];
    s[threadIdx.x] = a;
    __syncthreads();
    #pragma unroll
    for (int k = 128; k >= 1; k >>= 1) {
        if ((int)threadIdx.x < k) s[threadIdx.x] += s[threadIdx.x + k];
        __syncthreads();
    }
    if (threadIdx.x == 0)
        out[0] = 1.0f - s[0] / (float)(16 * 3 * IMG_H * IMG_W);
}

extern "C" void kernel_launch(void* const* d_in, const int* in_sizes, int n_in,
                              void* d_out, int out_size, void* d_ws, size_t ws_size,
                              hipStream_t stream) {
    const float* pred   = (const float*)d_in[0];
    const float* target = (const float*)d_in[1];
    const float* window = (const float*)d_in[2];

    v16h*  btab     = (v16h*)d_ws;                 // 32 * 32 B = 1 KB
    float* partials = (float*)d_ws + 256;          // 6144 floats = 24 KB
    float* out      = (float*)d_out;

    ssim_setup<<<1, 32, 0, stream>>>(window, btab);
    ssim_tile_kernel<<<N_BLOCKS, 256, 0, stream>>>(pred, target, btab, partials);
    ssim_finalize<<<1, 256, 0, stream>>>(partials, out);
}
